// BidirectionalRNN_39994735460585
// MI455X (gfx1250) — matrix operand order repaired and checked by reading, not compiled, not run on hardware
//
#include <hip/hip_runtime.h>
#include <hip/hip_bf16.h>

// ---------------------------------------------------------------------------
// BidirectionalRNN for MI455X (gfx1250, wave32, WMMA).
// Mixed precision: bf16 operands, f32 accumulation via v_wmma_f32_16x16x32_bf16.
// All GEMMs (embedding, X@Wx, recurrent h@Wh, final FC) use WMMA.
// ---------------------------------------------------------------------------

#define BB   128          // batch
#define TT   256          // time steps
#define HH   512          // hidden
#define RR   (BB * TT)    // total rows (t-major: row = t*BB + b)
#define KIN  300
#define KINP 320          // padded to multiple of 32
#define NC   45
#define NCP  64           // padded FC output

typedef unsigned short bf16_t;  // raw bf16 storage
typedef __attribute__((ext_vector_type(16))) __bf16 v16bf;
typedef __attribute__((ext_vector_type(8)))  float  v8f;

__device__ __forceinline__ bf16_t f2bf(float f) {
  union { float f; unsigned u; } x; x.f = f;
  unsigned r = x.u + 0x7FFFu + ((x.u >> 16) & 1u);   // round-to-nearest-even
  return (bf16_t)(r >> 16);
}

union FragU { uint4 q[2]; v16bf v; };

// A fragment: 16x32 (MxK) bf16, row-major source.
// lanes 0-15: M=lane, K(vgpr v) = {0..7,16..23}; lanes 16-31: K = {8..15,24..31}
__device__ __forceinline__ v16bf load_a_frag(const bf16_t* A, int lda,
                                             int mBase, int k, int lane) {
  const int m    = mBase + (lane & 15);
  const int half = lane >> 4;
  const bf16_t* p = A + (size_t)m * lda + k + half * 8;
  FragU f;
  f.q[0] = *(const uint4*)(p);        // K = half*8 + 0..7
  f.q[1] = *(const uint4*)(p + 16);   // K = half*8 + 16..23
  return f.v;
}

// B fragment: 32x16 (KxN) bf16, sourced from pre-transposed Bt[n][k] row-major.
// lanes 0-15 hold K={0..7,16..23} (col N=lane), lanes 16-31 hold K={8..15,24..31}
__device__ __forceinline__ v16bf load_b_frag(const bf16_t* Bt, int ldb,
                                             int nBase, int k, int lane) {
  const int n    = nBase + (lane & 15);
  const int half = lane >> 4;
  const bf16_t* p = Bt + (size_t)n * ldb + k + half * 8;
  FragU f;
  f.q[0] = *(const uint4*)(p);        // K = half*8 + 0..7
  f.q[1] = *(const uint4*)(p + 16);   // K = half*8 + 16..23
  return f.v;
}

// ---------------------------------------------------------------------------
// Generic bf16 WMMA GEMM:  C[M][N] = A[M][K] @ Bt[N][K]^T (+ bias) (+ addend)
// One wave computes a 16x64 tile (A fragment reused across 4 B tiles).
// All 5 fragment loads (10x global_load_b128) are issued before the first
// WMMA so the compiler can clause them and use partial s_wait_loadcnt values,
// keeping later loads in flight while earlier WMMAs execute.
// Requires M%16==0, N%64==0, K%32==0.  256 threads = 8 waves per block.
// ---------------------------------------------------------------------------
__global__ void wmma_gemm_bf16(const bf16_t* __restrict__ A, int lda,
                               const bf16_t* __restrict__ Bt, int ldb,
                               const float*  __restrict__ addend,   // [M][ldc] or null
                               const float*  __restrict__ bias,     // [N] or null
                               float*        __restrict__ Cf,       // [M][ldc] or null
                               bf16_t*       __restrict__ Cb,       // [M][ldc] or null
                               int M, int N, int K, int ldc) {
  const int lane    = threadIdx.x & 31;
  const int wave    = blockIdx.x * (blockDim.x >> 5) + (threadIdx.x >> 5);
  const int nGroups = N >> 6;
  const int total   = (M >> 4) * nGroups;
  if (wave >= total) return;                 // wave-uniform: EXEC stays all-1s

  const int mTile  = wave / nGroups;
  const int nGroup = wave - mTile * nGroups;
  const int mBase  = mTile << 4;
  const int nBase  = nGroup << 6;

  v8f acc[4] = {};
  for (int k = 0; k < K; k += 32) {
    if (k + 32 < K)  // prefetch next A slice -> global_prefetch_b8
      __builtin_prefetch(A + (size_t)(mBase + (lane & 15)) * lda + k + 32, 0, 1);
    // Issue ALL fragment loads first (one clause), then consume in order.
    v16bf a  = load_a_frag(A, lda, mBase, k, lane);
    v16bf b0 = load_b_frag(Bt, ldb, nBase +  0, k, lane);
    v16bf b1 = load_b_frag(Bt, ldb, nBase + 16, k, lane);
    v16bf b2 = load_b_frag(Bt, ldb, nBase + 32, k, lane);
    v16bf b3 = load_b_frag(Bt, ldb, nBase + 48, k, lane);
    acc[0] = __builtin_amdgcn_wmma_f32_16x16x32_bf16(false, a, false, b0, (short)0, acc[0], false, false);
    acc[1] = __builtin_amdgcn_wmma_f32_16x16x32_bf16(false, a, false, b1, (short)0, acc[1], false, false);
    acc[2] = __builtin_amdgcn_wmma_f32_16x16x32_bf16(false, a, false, b2, (short)0, acc[2], false, false);
    acc[3] = __builtin_amdgcn_wmma_f32_16x16x32_bf16(false, a, false, b3, (short)0, acc[3], false, false);
  }

  // C/D layout: vgpr r -> (M = mBase + (lane>=16)*8 + r, N = nBase + (lane&15))
  const int nCol = lane & 15;
  const int mOff = (lane >> 4) << 3;
#pragma unroll
  for (int j = 0; j < 4; ++j) {
    const int n = nBase + j * 16 + nCol;
    const float bv = bias ? bias[n] : 0.0f;
#pragma unroll
    for (int r = 0; r < 8; ++r) {
      const int m = mBase + mOff + r;
      float v = acc[j][r] + bv;
      if (addend) v += addend[(size_t)m * ldc + n];
      if (Cf) Cf[(size_t)m * ldc + n] = v;
      if (Cb) Cb[(size_t)m * ldc + n] = f2bf(v);
    }
  }
}

// ---------------------------------------------------------------------------
// LayerNorm + tanh over rows of H=512; one block (256 threads) per row.
// LDS tree reduction for mean/var.
// ---------------------------------------------------------------------------
__global__ void ln_tanh_kernel(const float* __restrict__ pre,
                               const float* __restrict__ g,
                               const float* __restrict__ b,
                               bf16_t* __restrict__ hout) {
  __shared__ float red[256];
  const int row = blockIdx.x;
  const int tid = threadIdx.x;
  const float* p = pre + (size_t)row * HH;
  float v0 = p[tid], v1 = p[tid + 256];

  red[tid] = v0 + v1;
  __syncthreads();
  for (int off = 128; off > 0; off >>= 1) {
    if (tid < off) red[tid] += red[tid + off];
    __syncthreads();
  }
  const float mean = red[0] * (1.0f / HH);
  __syncthreads();

  const float d0 = v0 - mean, d1 = v1 - mean;
  red[tid] = d0 * d0 + d1 * d1;
  __syncthreads();
  for (int off = 128; off > 0; off >>= 1) {
    if (tid < off) red[tid] += red[tid + off];
    __syncthreads();
  }
  const float inv = rsqrtf(red[0] * (1.0f / HH) + 1e-5f);

  const float y0 = tanhf(d0 * inv * g[tid] + b[tid]);
  const float y1 = tanhf(d1 * inv * g[tid + 256] + b[tid + 256]);
  hout[(size_t)row * HH + tid]       = f2bf(y0);
  hout[(size_t)row * HH + tid + 256] = f2bf(y1);
}

// ---------------------------------------------------------------------------
// Helpers: conversion / packing / gather kernels
// ---------------------------------------------------------------------------

// Wt[n][k] = bf16(W[k][n]), zero-padded to (Np, Kp)
__global__ void transpose_cvt(const float* __restrict__ W, bf16_t* __restrict__ Wt,
                              int K, int N, int Kp, int Np) {
  size_t i = (size_t)blockIdx.x * blockDim.x + threadIdx.x;
  size_t tot = (size_t)Np * Kp;
  if (i >= tot) return;
  int n = (int)(i / Kp), k = (int)(i - (size_t)n * Kp);
  float v = (n < N && k < K) ? W[(size_t)k * N + n] : 0.0f;
  Wt[i] = f2bf(v);
}

// x[b][t][c] (f32, c<KIN) -> xb[(t*BB+b)*KINP + c] (bf16, zero-padded)
__global__ void cvt_x_kernel(const float* __restrict__ x, bf16_t* __restrict__ xb) {
  size_t i = (size_t)blockIdx.x * blockDim.x + threadIdx.x;
  if (i >= (size_t)RR * KINP) return;
  int c = (int)(i % KINP);
  size_t row = i / KINP;
  int b = (int)(row % BB), t = (int)(row / BB);
  float v = (c < KIN) ? x[((size_t)b * TT + t) * KIN + c] : 0.0f;
  xb[i] = f2bf(v);
}

__global__ void vec_add_kernel(const float* a, const float* b, float* o, int n) {
  int i = blockIdx.x * blockDim.x + threadIdx.x;
  if (i < n) o[i] = a[i] + b[i];
}

__global__ void pad_copy_kernel(const float* src, float* dst, int n, int np) {
  int i = blockIdx.x * blockDim.x + threadIdx.x;
  if (i < np) dst[i] = (i < n) ? src[i] : 0.0f;
}

__global__ void fill_zero_bf16(bf16_t* p, int n) {
  int i = blockIdx.x * blockDim.x + threadIdx.x;
  if (i < n) p[i] = 0;
}

// concat[row][0:512] = l2r[row];  concat[row][512:1024] = r2l[idx(b,j)]
// row = j*BB + b ; idx = j < pad[b] ? pad[b]-j-1 : j.  uint4 = 8 bf16 per thread.
__global__ void concat_gather(const bf16_t* __restrict__ l2r,
                              const bf16_t* __restrict__ r2l,
                              const int* __restrict__ pad,
                              bf16_t* __restrict__ cc) {
  size_t i = (size_t)blockIdx.x * blockDim.x + threadIdx.x;
  if (i >= (size_t)RR * 128) return;      // 1024 cols / 8 per chunk
  int chunk = (int)(i % 128);
  size_t row = i / 128;
  int b = (int)(row % BB), j = (int)(row / BB);
  uint4 v;
  if (chunk < 64) {
    v = ((const uint4*)(l2r + row * HH))[chunk];
  } else {
    int p = pad[b];
    int idx = (j < p) ? (p - j - 1) : j;
    v = ((const uint4*)(r2l + ((size_t)idx * BB + b) * HH))[chunk - 64];
  }
  ((uint4*)(cc + row * 1024))[chunk] = v;
}

// tmp[(j*BB+b)][NCP] -> out[(b*TT+j)*NC + n], n < NC
__global__ void fc_store_kernel(const float* __restrict__ tmp, float* __restrict__ out) {
  size_t i = (size_t)blockIdx.x * blockDim.x + threadIdx.x;
  if (i >= (size_t)RR * NC) return;
  int n = (int)(i % NC);
  size_t row = i / NC;
  int b = (int)(row % BB), j = (int)(row / BB);
  out[((size_t)b * TT + j) * NC + n] = tmp[row * NCP + n];
}

// ---------------------------------------------------------------------------
static inline int gemm_blocks(int M, int N) {
  int waves = (M / 16) * (N / 64);
  return (waves + 7) / 8;                  // 8 waves (256 threads) per block
}

extern "C" void kernel_launch(void* const* d_in, const int* in_sizes, int n_in,
                              void* d_out, int out_size, void* d_ws, size_t ws_size,
                              hipStream_t stream) {
  (void)in_sizes; (void)n_in; (void)out_size;

  const float* x      = (const float*)d_in[0];
  const float* rx     = (const float*)d_in[1];
  const int*   pad    = (const int*)  d_in[2];
  const float* W_emb  = (const float*)d_in[4];
  const float* b_emb  = (const float*)d_in[5];
  const float* Wx[2]  = {(const float*)d_in[6],  (const float*)d_in[10]};
  const float* bx[2]  = {(const float*)d_in[7],  (const float*)d_in[11]};
  const float* Wh[2]  = {(const float*)d_in[8],  (const float*)d_in[12]};
  const float* bh[2]  = {(const float*)d_in[9],  (const float*)d_in[13]};
  const float* ln_g   = (const float*)d_in[14];
  const float* ln_b   = (const float*)d_in[15];
  const float* W_fc   = (const float*)d_in[16];
  const float* b_fc   = (const float*)d_in[17];
  float* out = (float*)d_out;

  // -------- workspace layout (256B aligned blocks) --------
  char* base = (char*)d_ws;
  size_t off = 0;
  auto take = [&](size_t bytes) -> char* {
    char* p = base + off;
    off += (bytes + 255) & ~(size_t)255;
    return p;
  };
  bf16_t* WembT  = (bf16_t*)take((size_t)HH * KINP * 2);          // [512][320]
  bf16_t* WxT[2][3], *WhT[2][3];
  for (int d = 0; d < 2; ++d)
    for (int l = 0; l < 3; ++l) {
      WxT[d][l] = (bf16_t*)take((size_t)HH * HH * 2);
      WhT[d][l] = (bf16_t*)take((size_t)HH * HH * 2);
    }
  bf16_t* WfcT   = (bf16_t*)take((size_t)NCP * 1024 * 2);         // [64][1024]
  float*  bsum   = (float*) take((size_t)6 * HH * 4);
  float*  bfcp   = (float*) take((size_t)NCP * 4);
  bf16_t* xb     = (bf16_t*)take((size_t)RR * KINP * 2);
  bf16_t* ebuf   = (bf16_t*)take((size_t)RR * HH * 2);
  bf16_t* ping   = (bf16_t*)take((size_t)RR * HH * 2);
  bf16_t* pong   = (bf16_t*)take((size_t)RR * HH * 2);
  bf16_t* finL   = (bf16_t*)take((size_t)RR * HH * 2);
  bf16_t* finR   = (bf16_t*)take((size_t)RR * HH * 2);
  float*  XW     = (float*) take((size_t)RR * HH * 4);            // reused as concat
  float*  fc_tmp = (float*) take((size_t)RR * NCP * 4);
  float*  preact = (float*) take((size_t)BB * HH * 4);
  bf16_t* h0     = (bf16_t*)take((size_t)BB * HH * 2);
  if (off > ws_size) return;  // insufficient scratch: do nothing (deterministic)

  const int TPB = 256;
  auto blks = [&](size_t n) -> int { return (int)((n + TPB - 1) / TPB); };

  // -------- one-time weight packing (bf16 + transpose) --------
  transpose_cvt<<<blks((size_t)HH * KINP), TPB, 0, stream>>>(W_emb, WembT, KIN, HH, KINP, HH);
  for (int d = 0; d < 2; ++d)
    for (int l = 0; l < 3; ++l) {
      transpose_cvt<<<blks((size_t)HH * HH), TPB, 0, stream>>>(Wx[d] + (size_t)l * HH * HH,
                                                              WxT[d][l], HH, HH, HH, HH);
      transpose_cvt<<<blks((size_t)HH * HH), TPB, 0, stream>>>(Wh[d] + (size_t)l * HH * HH,
                                                              WhT[d][l], HH, HH, HH, HH);
      vec_add_kernel<<<blks(HH), TPB, 0, stream>>>(bx[d] + l * HH, bh[d] + l * HH,
                                                   bsum + (d * 3 + l) * HH, HH);
    }
  transpose_cvt<<<blks((size_t)NCP * 1024), TPB, 0, stream>>>(W_fc, WfcT, 1024, NC, 1024, NCP);
  pad_copy_kernel<<<1, TPB, 0, stream>>>(b_fc, bfcp, NC, NCP);
  fill_zero_bf16<<<blks((size_t)BB * HH), TPB, 0, stream>>>(h0, BB * HH);

  // -------- per direction --------
  for (int d = 0; d < 2; ++d) {
    // embed: ebuf = bf16( xb @ WembT^T + b_emb ),  K = 320
    cvt_x_kernel<<<blks((size_t)RR * KINP), TPB, 0, stream>>>(d ? rx : x, xb);
    wmma_gemm_bf16<<<gemm_blocks(RR, HH), TPB, 0, stream>>>(
        xb, KINP, WembT, KINP, nullptr, b_emb, nullptr, ebuf, RR, HH, KINP, HH);

    for (int l = 0; l < 3; ++l) {
      const bf16_t* inbuf  = (l == 0) ? ebuf : (l == 1 ? ping : pong);
      bf16_t*       outbuf = (l == 0) ? ping : (l == 1 ? pong : (d ? finR : finL));

      // XW[t*BB+b][:] = inbuf @ Wx  (all timesteps, one big WMMA GEMM)
      wmma_gemm_bf16<<<gemm_blocks(RR, HH), TPB, 0, stream>>>(
          inbuf, HH, WxT[d][l], HH, nullptr, nullptr, XW, nullptr, RR, HH, HH, HH);

      // sequential scan over T: preact = h_{t-1}@Wh + XW_t + (bx+bh); LN; tanh
      for (int t = 0; t < TT; ++t) {
        const bf16_t* hprev = t ? outbuf + (size_t)(t - 1) * BB * HH : h0;
        wmma_gemm_bf16<<<gemm_blocks(BB, HH), TPB, 0, stream>>>(
            hprev, HH, WhT[d][l], HH, XW + (size_t)t * BB * HH,
            bsum + (d * 3 + l) * HH, preact, nullptr, BB, HH, HH, HH);
        ln_tanh_kernel<<<BB, TPB, 0, stream>>>(
            preact, ln_g + l * HH, ln_b + l * HH, outbuf + (size_t)t * BB * HH);
      }
    }
  }

  // -------- gather reversed dir, concat, FC --------
  bf16_t* cc = (bf16_t*)XW;  // XW is free now; R*512*4 bytes == R*1024*2 bytes
  concat_gather<<<blks((size_t)RR * 128), TPB, 0, stream>>>(finL, finR, pad, cc);
  wmma_gemm_bf16<<<gemm_blocks(RR, NCP), TPB, 0, stream>>>(
      cc, 1024, WfcT, 1024, nullptr, bfcp, fc_tmp, nullptr, RR, NCP, 1024, NCP);
  fc_store_kernel<<<blks((size_t)RR * NC), TPB, 0, stream>>>(fc_tmp, out);
}
